// MessagePassing_18872086298992
// MI455X (gfx1250) — matrix operand order repaired and checked by reading, not compiled, not run on hardware
//
#include <hip/hip_runtime.h>
#include <hip/hip_bf16.h>

// ---------------------------------------------------------------------------
// MessagePassing layer for MI455X (gfx1250).
//   - bf16 WMMA (v_wmma_f32_16x16x32_bf16) for both MLPs, f32 accumulate
//   - x pre-converted to a bf16 copy (L2-resident) so endpoint gathers go
//     straight cache->LDS via global_load_async_to_lds_b128 (ASYNCcnt path)
//   - scatter-add with f32 global atomics (L2-resident agg)
// ---------------------------------------------------------------------------

#define NODE_DIM 64
#define EDGE_DIM 64
#define N_NODES  100000
#define N_EDGES  1600000
#define DIN_E    192      // 2*NODE_DIM + EDGE_DIM
#define DIN_V    128      // NODE_DIM + EDGE_DIM

typedef __attribute__((ext_vector_type(16))) __bf16 v16bf;
typedef __attribute__((ext_vector_type(8)))  __bf16 v8bf;
typedef __attribute__((ext_vector_type(8)))  float  v8f;

union AV { v16bf v; v8bf h[2]; };

// float -> bf16 (round-to-nearest-even)
__device__ __forceinline__ unsigned bfbits(float f) {
  unsigned u = __builtin_bit_cast(unsigned, f);
  return (u + 0x7FFFu + ((u >> 16) & 1u)) >> 16;
}
__device__ __forceinline__ __bf16 f2bf(float f) {
  unsigned short s = (unsigned short)bfbits(f);
  return __builtin_bit_cast(__bf16, s);
}
__device__ __forceinline__ unsigned pack2(float a, float b) {
  return bfbits(a) | (bfbits(b) << 16);
}
// store 4 floats as 4 bf16 (one 8-byte LDS store)
__device__ __forceinline__ void st4bf(__bf16* p, float4 f) {
  uint2 u;
  u.x = pack2(f.x, f.y);
  u.y = pack2(f.z, f.w);
  *(uint2*)p = u;
}
__device__ __forceinline__ float gelu_exact(float v) {
  return 0.5f * v * (1.0f + erff(v * 0.70710678118654752440f));
}
// LDS byte address of a __shared__ object (generic ptr low 32 bits)
__device__ __forceinline__ unsigned lds_addr(const void* p) {
  return (unsigned)(unsigned long long)p;
}
// async 16B copy: global (SADDR base + 32-bit offset) -> LDS
__device__ __forceinline__ void async_b128(unsigned lds, unsigned goff,
                                           const void* base) {
  asm volatile("global_load_async_to_lds_b128 %0, %1, %2"
               :: "v"(lds), "v"(goff), "s"(base) : "memory");
}
__device__ __forceinline__ void wait_async0() {
  asm volatile("s_wait_asynccnt 0x0" ::: "memory");
}

// ---------------------------------------------------------------------------
// prep: zero agg and build bf16 copy of x (both L2-resident afterwards)
// ---------------------------------------------------------------------------
__global__ void mp_prep_kernel(const float* __restrict__ x,
                               __bf16* __restrict__ xbf,
                               float* __restrict__ agg) {
  const long long n2 = (long long)N_NODES * NODE_DIM / 2;  // pairs
  long long i = (long long)blockIdx.x * blockDim.x + threadIdx.x;
  long long stride = (long long)gridDim.x * blockDim.x;
  for (; i < n2; i += stride) {
    ((unsigned*)xbf)[i] = pack2(x[2 * i], x[2 * i + 1]);
    ((float2*)agg)[i] = make_float2(0.0f, 0.0f);
  }
}

// ---------------------------------------------------------------------------
// Edge MLP + scatter-add.  4 waves/block, 1 tile of 16 edges per wave.
// ---------------------------------------------------------------------------
#define NWAVE_E 4
#define EPADA  200   // padded LDS row (elements) for A (DIN_E=192 + 8)
#define EPAD2  72    // padded LDS row for 64-wide weights

__launch_bounds__(128)
__global__ void mp_edge_kernel(const __bf16* __restrict__ xbf,
                               const long long* __restrict__ ei,
                               const float* __restrict__ ef,
                               const float* __restrict__ We1,
                               const float* __restrict__ be1,
                               const float* __restrict__ We2,
                               const float* __restrict__ be2,
                               float* __restrict__ agg) {
  __shared__ __align__(16) __bf16 sW1t[64][EPADA];          // We1^T : [n][k]
  __shared__ __align__(16) __bf16 sW2t[64][EPAD2];          // We2^T : [n][k]
  __shared__ __align__(16) __bf16 sA[NWAVE_E][16][EPADA];   // staged A tiles
  __shared__ int   sDst[NWAVE_E][16];
  __shared__ float sB1[64], sB2[64];

  const int tid = threadIdx.x;

  // load + transpose weights into LDS (coalesced global reads)
  for (int idx = tid; idx < DIN_E * 64; idx += blockDim.x) {
    int k = idx >> 6, n = idx & 63;
    sW1t[n][k] = f2bf(We1[idx]);
  }
  for (int idx = tid; idx < 64 * 64; idx += blockDim.x) {
    int k = idx >> 6, n = idx & 63;
    sW2t[n][k] = f2bf(We2[idx]);
  }
  if (tid < 64) { sB1[tid] = be1[tid]; sB2[tid] = be2[tid]; }
  __syncthreads();

  const int wave  = tid >> 5;
  const int lane  = tid & 31;
  const int khalf = lane >> 4;      // K-half select (ISA A/B fragment layout)
  const int nlo   = lane & 15;
  const long long tile = (long long)blockIdx.x * NWAVE_E + wave; // < 100000
  const long long e0   = tile * 16;

  // ---- stage A = [x[src] | x[dst] | ef] as bf16 ----
  // x rows (already bf16): async cache->LDS, 8 lanes x 16B per 128B row.
  {
    const int c  = lane & 7;     // 16B chunk within a row
    const int rb = lane >> 3;    // 0..3
    long long sIdx[4], dIdx[4];
#pragma unroll
    for (int t = 0; t < 4; ++t) {
      const int m = rb + 4 * t;
      sIdx[t] = ei[e0 + m];
      dIdx[t] = ei[(long long)N_EDGES + e0 + m];
      if (c == 0) sDst[wave][m] = (int)dIdx[t];
    }
#pragma unroll
    for (int t = 0; t < 4; ++t) {
      const int m = rb + 4 * t;
      const unsigned lds_s = lds_addr(&sA[wave][m][0]) + (unsigned)(c * 16);
      async_b128(lds_s,        (unsigned)(sIdx[t] * 128 + c * 16), xbf);
      async_b128(lds_s + 128u, (unsigned)(dIdx[t] * 128 + c * 16), xbf);
    }
    // ef (f32 stream from HBM) -> bf16 cols 128..191, overlapped with async
    const int m2 = lane >> 1;
    const int h  = lane & 1;
    const float4* re = (const float4*)(ef + (e0 + m2) * EDGE_DIM);
#pragma unroll
    for (int j = 0; j < 8; ++j)
      st4bf(&sA[wave][m2][128 + h * 32 + j * 4], re[h * 8 + j]);
    wait_async0();
  }
  __builtin_amdgcn_wave_barrier();   // DS ops are in-order within the wave

  const __bf16* Arow = sA[wave][lane & 15];   // A: lane = matrix row M

  // ---- layer 1: [16x192] x [192x64] -> 24 WMMAs ----
  v8f acc[4] = {};
#pragma unroll
  for (int kb = 0; kb < DIN_E; kb += 32) {
    AV a;
    a.h[0] = *(const v8bf*)(Arow + kb +      8 * khalf);
    a.h[1] = *(const v8bf*)(Arow + kb + 16 + 8 * khalf);
#pragma unroll
    for (int nb = 0; nb < 4; ++nb) {
      AV b;
      const __bf16* Brow = sW1t[nb * 16 + nlo];
      b.h[0] = *(const v8bf*)(Brow + kb +      8 * khalf);
      b.h[1] = *(const v8bf*)(Brow + kb + 16 + 8 * khalf);
      acc[nb] = __builtin_amdgcn_wmma_f32_16x16x32_bf16(
          false, a.v, false, b.v, (short)0, acc[nb], false, false);
    }
  }

  // ---- bias + exact GELU, restage as bf16 A tile (cols 0..63) ----
#pragma unroll
  for (int nb = 0; nb < 4; ++nb) {
    const int n = nb * 16 + nlo;
    const float bias = sB1[n];
#pragma unroll
    for (int r = 0; r < 8; ++r) {
      float v = acc[nb][r] + bias;            // C layout: M = r + 8*khalf
      sA[wave][r + 8 * khalf][n] = f2bf(gelu_exact(v));
    }
  }
  __builtin_amdgcn_wave_barrier();

  // ---- layer 2: [16x64] x [64x64] -> 8 WMMAs ----
  v8f acc2[4] = {};
#pragma unroll
  for (int kb = 0; kb < 64; kb += 32) {
    AV a;
    a.h[0] = *(const v8bf*)(Arow + kb +      8 * khalf);
    a.h[1] = *(const v8bf*)(Arow + kb + 16 + 8 * khalf);
#pragma unroll
    for (int nb = 0; nb < 4; ++nb) {
      AV b;
      const __bf16* Brow = sW2t[nb * 16 + nlo];
      b.h[0] = *(const v8bf*)(Brow + kb +      8 * khalf);
      b.h[1] = *(const v8bf*)(Brow + kb + 16 + 8 * khalf);
      acc2[nb] = __builtin_amdgcn_wmma_f32_16x16x32_bf16(
          false, a.v, false, b.v, (short)0, acc2[nb], false, false);
    }
  }

  // ---- bias + scatter-add e' into agg[dst] (L2-resident f32 atomics) ----
#pragma unroll
  for (int nb = 0; nb < 4; ++nb) {
    const int n = nb * 16 + nlo;
    const float bias = sB2[n];
#pragma unroll
    for (int r = 0; r < 8; ++r) {
      float v = acc2[nb][r] + bias;
      long long row = sDst[wave][r + 8 * khalf];
      unsafeAtomicAdd(agg + row * NODE_DIM + n, v);
    }
  }
}

// ---------------------------------------------------------------------------
// Node MLP + residual.  8 waves/block, 1 tile of 16 nodes per wave.
// ---------------------------------------------------------------------------
#define NWAVE_V 8
#define VPADA   136  // DIN_V=128 + 8
#define N_TILES_V (N_NODES / 16)   // 6250

__launch_bounds__(256)
__global__ void mp_node_kernel(const float* __restrict__ x,
                               const __bf16* __restrict__ xbf,
                               const float* __restrict__ agg,
                               const float* __restrict__ Wv1,
                               const float* __restrict__ bv1,
                               const float* __restrict__ Wv2,
                               const float* __restrict__ bv2,
                               float* __restrict__ out) {
  __shared__ __align__(16) __bf16 sW1t[64][VPADA];
  __shared__ __align__(16) __bf16 sW2t[64][EPAD2];
  __shared__ __align__(16) __bf16 sA[NWAVE_V][16][VPADA];
  __shared__ float sB1[64], sB2[64];

  const int tid = threadIdx.x;
  for (int idx = tid; idx < DIN_V * 64; idx += blockDim.x) {
    int k = idx >> 6, n = idx & 63;
    sW1t[n][k] = f2bf(Wv1[idx]);
  }
  for (int idx = tid; idx < 64 * 64; idx += blockDim.x) {
    int k = idx >> 6, n = idx & 63;
    sW2t[n][k] = f2bf(Wv2[idx]);
  }
  if (tid < 64) { sB1[tid] = bv1[tid]; sB2[tid] = bv2[tid]; }
  __syncthreads();

  const int wave  = tid >> 5;
  const int lane  = tid & 31;
  const int khalf = lane >> 4;
  const int nlo   = lane & 15;
  const long long tile = (long long)blockIdx.x * NWAVE_V + wave;
  if (tile >= N_TILES_V) return;
  const long long i0 = tile * 16;

  // ---- stage A = [x | agg]: x rows async from bf16 copy, agg converted ----
  {
    const int c  = lane & 7;
    const int rb = lane >> 3;
#pragma unroll
    for (int t = 0; t < 4; ++t) {
      const int m = rb + 4 * t;
      const unsigned lds_x = lds_addr(&sA[wave][m][0]) + (unsigned)(c * 16);
      async_b128(lds_x, (unsigned)((i0 + m) * 128 + c * 16), xbf);
    }
    const int m2 = lane >> 1;
    const int h  = lane & 1;
    const float4* rg = (const float4*)(agg + (i0 + m2) * NODE_DIM);
#pragma unroll
    for (int j = 0; j < 8; ++j)
      st4bf(&sA[wave][m2][64 + h * 32 + j * 4], rg[h * 8 + j]);
    wait_async0();
  }
  __builtin_amdgcn_wave_barrier();

  const __bf16* Arow = sA[wave][lane & 15];

  // ---- layer 1: [16x128] x [128x64] -> 16 WMMAs ----
  v8f acc[4] = {};
#pragma unroll
  for (int kb = 0; kb < DIN_V; kb += 32) {
    AV a;
    a.h[0] = *(const v8bf*)(Arow + kb +      8 * khalf);
    a.h[1] = *(const v8bf*)(Arow + kb + 16 + 8 * khalf);
#pragma unroll
    for (int nb = 0; nb < 4; ++nb) {
      AV b;
      const __bf16* Brow = sW1t[nb * 16 + nlo];
      b.h[0] = *(const v8bf*)(Brow + kb +      8 * khalf);
      b.h[1] = *(const v8bf*)(Brow + kb + 16 + 8 * khalf);
      acc[nb] = __builtin_amdgcn_wmma_f32_16x16x32_bf16(
          false, a.v, false, b.v, (short)0, acc[nb], false, false);
    }
  }

#pragma unroll
  for (int nb = 0; nb < 4; ++nb) {
    const int n = nb * 16 + nlo;
    const float bias = sB1[n];
#pragma unroll
    for (int r = 0; r < 8; ++r) {
      float v = acc[nb][r] + bias;
      sA[wave][r + 8 * khalf][n] = f2bf(gelu_exact(v));
    }
  }
  __builtin_amdgcn_wave_barrier();

  // ---- layer 2: [16x64] x [64x64] -> 8 WMMAs ----
  v8f acc2[4] = {};
#pragma unroll
  for (int kb = 0; kb < 64; kb += 32) {
    AV a;
    a.h[0] = *(const v8bf*)(Arow + kb +      8 * khalf);
    a.h[1] = *(const v8bf*)(Arow + kb + 16 + 8 * khalf);
#pragma unroll
    for (int nb = 0; nb < 4; ++nb) {
      AV b;
      const __bf16* Brow = sW2t[nb * 16 + nlo];
      b.h[0] = *(const v8bf*)(Brow + kb +      8 * khalf);
      b.h[1] = *(const v8bf*)(Brow + kb + 16 + 8 * khalf);
      acc2[nb] = __builtin_amdgcn_wmma_f32_16x16x32_bf16(
          false, a.v, false, b.v, (short)0, acc2[nb], false, false);
    }
  }

  // ---- residual: out = x + v' ----
#pragma unroll
  for (int nb = 0; nb < 4; ++nb) {
    const int n = nb * 16 + nlo;
    const float bias = sB2[n];
#pragma unroll
    for (int r = 0; r < 8; ++r) {
      long long row = i0 + r + 8 * khalf;
      float v = acc2[nb][r] + bias + x[row * NODE_DIM + n];
      out[row * NODE_DIM + n] = v;
    }
  }
}

// ---------------------------------------------------------------------------
extern "C" void kernel_launch(void* const* d_in, const int* in_sizes, int n_in,
                              void* d_out, int out_size, void* d_ws, size_t ws_size,
                              hipStream_t stream) {
  const float*      x   = (const float*)d_in[0];
  const long long*  ei  = (const long long*)d_in[1];
  const float*      ef  = (const float*)d_in[2];
  const float*      We1 = (const float*)d_in[3];
  const float*      be1 = (const float*)d_in[4];
  const float*      We2 = (const float*)d_in[5];
  const float*      be2 = (const float*)d_in[6];
  const float*      Wv1 = (const float*)d_in[7];
  const float*      bv1 = (const float*)d_in[8];
  const float*      Wv2 = (const float*)d_in[9];
  const float*      bv2 = (const float*)d_in[10];
  float* out = (float*)d_out;

  // workspace layout: agg (f32, 25.6MB) | x_bf16 (12.8MB)
  float*  agg = (float*)d_ws;
  __bf16* xbf = (__bf16*)((char*)d_ws +
                          (size_t)N_NODES * NODE_DIM * sizeof(float));

  // 1) zero agg + convert x to bf16 (both stay L2-resident)
  mp_prep_kernel<<<4096, 256, 0, stream>>>(x, xbf, agg);

  // 2) edge MLP + scatter: 100000 tiles, 4 waves/block
  mp_edge_kernel<<<N_EDGES / (16 * NWAVE_E), 128, 0, stream>>>(
      xbf, ei, ef, We1, be1, We2, be2, agg);

  // 3) node MLP + residual: 6250 tiles, 8 waves/block
  mp_node_kernel<<<(N_TILES_V + NWAVE_V - 1) / NWAVE_V, 256, 0, stream>>>(
      x, xbf, agg, Wv1, bv1, Wv2, bv2, out);
}
